// AxialBlock_58600533786747
// MI455X (gfx1250) — matrix operand order repaired and checked by reading, not compiled, not run on hardware
//
#include <hip/hip_runtime.h>
#include <hip/hip_bf16.h>

// Problem constants (from reference): T=4, E=4096, B=2, D=128, N=384
#define TT 4
#define EE 4096
#define BB 2
#define DD 128
#define NN 384
#define EBD (EE * BB * DD)       // 1,048,576
#define TEBD (TT * EBD)          // 4,194,304
#define FEAT_ROWS (NN * NN * BB) // 294,912

typedef __attribute__((ext_vector_type(2))) float v2f;
typedef __attribute__((ext_vector_type(8))) float v8f;
typedef __attribute__((ext_vector_type(16))) __bf16 v16bf;
typedef __attribute__((ext_vector_type(16))) unsigned short v16u;

// ---------------------------------------------------------------------------
// f32 WMMA GEMM (K=128, Nout=128):  out[r,o] = sum_k in[r,k]*W[o,k] + bias[o]
// Used for the small GEMMs (xg, xadd). One wave = 16-row x 128-col tile.
// ---------------------------------------------------------------------------
__global__ __launch_bounds__(256) void wmma_gemm_f32(
    const float* __restrict__ in, const float* __restrict__ W, int lda,
    const float* __restrict__ bias, float* __restrict__ out) {
  const int wave = blockIdx.x * (blockDim.x >> 5) + (threadIdx.x >> 5);
  const int lane = threadIdx.x & 31;
  const int hi = lane >> 4;
  const int lo = lane & 15;
  const int row0 = wave * 16;

  const float* ap = in + (size_t)(row0 + lo) * 128 + 2 * hi;

  v8f acc[8];
#pragma unroll
  for (int j = 0; j < 8; ++j) acc[j] = (v8f){0.f,0.f,0.f,0.f,0.f,0.f,0.f,0.f};

#pragma unroll 4
  for (int k0 = 0; k0 < 128; k0 += 4) {
    v2f a = *(const v2f*)(ap + k0);
#pragma unroll
    for (int j = 0; j < 8; ++j) {
      v2f b = *(const v2f*)(W + (size_t)(j * 16 + lo) * lda + k0 + 2 * hi);
      acc[j] = __builtin_amdgcn_wmma_f32_16x16x4_f32(
          false, a, false, b, (short)0, acc[j], false, false);
    }
  }

#pragma unroll
  for (int j = 0; j < 8; ++j) {
    const int o = j * 16 + lo;
    const float bv = bias[o];
#pragma unroll
    for (int v = 0; v < 8; ++v) {
      const int row = row0 + v + 8 * hi;
      out[(size_t)row * 128 + o] = acc[j][v] + bv;
    }
  }
}

// ---------------------------------------------------------------------------
// Split-bf16 (bf16x3) WMMA GEMM for the dominant feats GEMM.
//   out[r,o] = sum_k in[r,k]*W1[o,k] + (eidx>=0 ? xadd[e,b,o] : padvec[o])
// f32 operands split as hi+lo truncated bf16; D = Ahi*Bhi + Ahi*Blo + Alo*Bhi
// via v_wmma_f32_16x16x32_bf16 (8x the MACs/instr of the f32 K=4 op),
// pushing this GEMM from wmma-issue-bound down to the 23.3 TB/s HBM roofline.
// A-frag (16-bit A 16x32): lane K = hi*16+{0..15} -> two
// contiguous 32B f32 segments. B-frag (16-bit B 32x16): lane K = hi*16+{0..15}
// at column o -> one contiguous 32B load from row-major bf16 [o][k].
// ---------------------------------------------------------------------------
__global__ __launch_bounds__(256) void wmma_gemm_bf16x3(
    const float* __restrict__ in, const unsigned short* __restrict__ Whi,
    const unsigned short* __restrict__ Wlo, const int* __restrict__ eidx,
    const float* __restrict__ xadd, const float* __restrict__ padvec,
    float* __restrict__ out) {
  const int wave = blockIdx.x * (blockDim.x >> 5) + (threadIdx.x >> 5);
  const int lane = threadIdx.x & 31;
  const int hi = lane >> 4;
  const int lo = lane & 15;
  const int row0 = wave * 16;

  const float* arow = in + (size_t)(row0 + lo) * 128;

  v8f acc[8];
#pragma unroll
  for (int j = 0; j < 8; ++j) acc[j] = (v8f){0.f,0.f,0.f,0.f,0.f,0.f,0.f,0.f};

#pragma unroll
  for (int c = 0; c < 4; ++c) {
    const int k0 = 32 * c;
    // A fragment: 16 f32 from two contiguous 32B segments
    v8f af0 = *(const v8f*)(arow + k0 + hi * 16);
    v8f af1 = *(const v8f*)(arow + k0 + hi * 16 + 8);
    v16u uhi, ulo;
#pragma unroll
    for (int i = 0; i < 16; ++i) {
      const float x = (i < 8) ? af0[i] : af1[i - 8];
      const unsigned bits = __float_as_uint(x);
      const float fh = __uint_as_float(bits & 0xFFFF0000u);
      const float rl = x - fh;
      uhi[i] = (unsigned short)(bits >> 16);
      ulo[i] = (unsigned short)(__float_as_uint(rl) >> 16);
    }
    const v16bf ahi = __builtin_bit_cast(v16bf, uhi);
    const v16bf alo = __builtin_bit_cast(v16bf, ulo);

#pragma unroll
    for (int j = 0; j < 8; ++j) {
      const int o = j * 16 + lo;
      const v16bf bhi = *(const v16bf*)(Whi + (size_t)o * 128 + k0 + hi * 16);
      const v16bf blo = *(const v16bf*)(Wlo + (size_t)o * 128 + k0 + hi * 16);
      acc[j] = __builtin_amdgcn_wmma_f32_16x16x32_bf16(
          false, ahi, false, bhi, (short)0, acc[j], false, false);
      acc[j] = __builtin_amdgcn_wmma_f32_16x16x32_bf16(
          false, ahi, false, blo, (short)0, acc[j], false, false);
      acc[j] = __builtin_amdgcn_wmma_f32_16x16x32_bf16(
          false, alo, false, bhi, (short)0, acc[j], false, false);
    }
  }

  // Epilogue: per-row lookup add (scatter-cell -> xadd row, else pad_vec)
  const float* addp[8];
#pragma unroll
  for (int v = 0; v < 8; ++v) {
    const int row = row0 + v + 8 * hi;
    const int nm = row >> 1;  // B == 2
    const int b = row & 1;
    const int e = eidx[nm];
    addp[v] = (e >= 0) ? (xadd + ((size_t)e * 2 + b) * 128) : padvec;
  }
#pragma unroll
  for (int j = 0; j < 8; ++j) {
    const int o = j * 16 + lo;
#pragma unroll
    for (int v = 0; v < 8; ++v) {
      const int row = row0 + v + 8 * hi;
      out[(size_t)row * 128 + o] = acc[j][v] + addp[v][o];
    }
  }
}

// W_eff[o2, c] = sum_o W_f[o2, o] * W_lin[o, c]   (128 x 256)
__global__ void k_weff(const float* __restrict__ W_f,
                       const float* __restrict__ W_lin,
                       float* __restrict__ W_eff) {
  const int idx = blockIdx.x * blockDim.x + threadIdx.x;  // 32768 threads
  const int o2 = idx >> 8;
  const int c = idx & 255;
  float s = 0.f;
#pragma unroll 4
  for (int o = 0; o < 128; ++o) s += W_f[o2 * 128 + o] * W_lin[o * 256 + c];
  W_eff[o2 * 256 + c] = s;
}

// Split W1 = W_eff[:,0:128] into truncated-bf16 hi/lo, row-major [o][k]
__global__ void k_wsplit(const float* __restrict__ W_eff,
                         unsigned short* __restrict__ Whi,
                         unsigned short* __restrict__ Wlo) {
  const int idx = blockIdx.x * blockDim.x + threadIdx.x;  // 16384 threads
  const int o = idx >> 7;
  const int k = idx & 127;
  const float w = W_eff[o * 256 + k];
  const unsigned bits = __float_as_uint(w);
  const float fh = __uint_as_float(bits & 0xFFFF0000u);
  Whi[idx] = (unsigned short)(bits >> 16);
  Wlo[idx] = (unsigned short)(__float_as_uint(w - fh) >> 16);
}

// b_eff[o2] = W_f @ b_lin + b_f ;  pad_vec[o2] = pad * rowsum(W2[o2,:]) + b_eff
__global__ void k_beff_pad(const float* __restrict__ W_f,
                           const float* __restrict__ b_lin,
                           const float* __restrict__ b_f,
                           const float* __restrict__ pad,
                           const float* __restrict__ W_eff,
                           float* __restrict__ b_eff,
                           float* __restrict__ pad_vec) {
  const int o2 = threadIdx.x;  // 128 threads
  float be = b_f[o2];
  for (int o = 0; o < 128; ++o) be += W_f[o2 * 128 + o] * b_lin[o];
  float s = 0.f;
  for (int c = 0; c < 128; ++c) s += W_eff[o2 * 256 + 128 + c];
  b_eff[o2] = be;
  pad_vec[o2] = pad[0] * s + be;
}

// xmean over T of xg (stored in d_out's x region)
__global__ void k_xmean(const float* __restrict__ xg, float* __restrict__ xm) {
  const int i = blockIdx.x * blockDim.x + threadIdx.x;  // EBD threads
  float s = 0.f;
#pragma unroll
  for (int t = 0; t < TT; ++t) s += xg[(size_t)t * EBD + i];
  xm[i] = s * (1.0f / TT);
}

__global__ void k_eidx_fill(int* __restrict__ eidx) {
  const int i = blockIdx.x * blockDim.x + threadIdx.x;
  if (i < NN * NN) eidx[i] = -1;
}

__global__ void k_eidx_scatter(const int* __restrict__ nodes,
                               int* __restrict__ eidx) {
  const int e = blockIdx.x * blockDim.x + threadIdx.x;
  if (e >= EE) return;
  const int n0 = nodes[2 * e] - 1;  // nodes in [1, N]
  const int n1 = nodes[2 * e + 1] - 1;
  if (n0 >= 0 && n0 < NN && n1 >= 0 && n1 < NN) eidx[n0 * NN + n1] = e;
}

// x_out[t,e,b,:] += feats_out[(n0-1)*N + (n1-1), b, :]
__global__ void k_final(const int* __restrict__ nodes,
                        const float* __restrict__ feats_out,
                        float* __restrict__ x_out) {
  const int i = blockIdx.x * blockDim.x + threadIdx.x;  // TEBD threads
  const int ebd = i & (EBD - 1);
  const int e = ebd >> 8;
  const int b = (ebd >> 7) & 1;
  const int o = ebd & 127;
  int n0 = nodes[2 * e] - 1; if (n0 < 0) n0 = 0;
  int n1 = nodes[2 * e + 1] - 1; if (n1 < 0) n1 = 0;
  const size_t row = ((size_t)n0 * NN + n1) * BB + b;
  x_out[i] += feats_out[row * 128 + o];
}

extern "C" void kernel_launch(void* const* d_in, const int* in_sizes, int n_in,
                              void* d_out, int out_size, void* d_ws,
                              size_t ws_size, hipStream_t stream) {
  const float* x = (const float*)d_in[0];
  const float* feats = (const float*)d_in[1];
  const int* nodes = (const int*)d_in[2];
  // d_in[3], d_in[4]: padding masks (unused by reference math)
  const float* pad = (const float*)d_in[5];
  const float* W_g = (const float*)d_in[6];
  const float* b_g = (const float*)d_in[7];
  const float* W_lin = (const float*)d_in[8];
  const float* b_lin = (const float*)d_in[9];
  const float* W_f = (const float*)d_in[10];
  const float* b_f = (const float*)d_in[11];

  float* out_x = (float*)d_out;             // (T,E,B,D)
  float* out_feats = (float*)d_out + TEBD;  // (N,N,B,D)

  // Workspace layout (floats)
  float* ws = (float*)d_ws;
  float* W_eff = ws;                          // 32768
  float* b_eff = ws + 32768;                  // 128
  float* pad_vec = ws + 32896;                // 128
  float* xmean = ws + 33024;                  // EBD
  float* xadd = xmean + EBD;                  // E*B*128
  int* eidx = (int*)(xadd + EE * BB * 128);   // N*N ints (147456)
  unsigned short* Whi = (unsigned short*)(eidx + NN * NN);  // 16384 u16
  unsigned short* Wlo = Whi + 128 * 128;                    // 16384 u16

  // 1) Folded weight W_eff = W_f @ W_lin, then b_eff / pad_vec / bf16 split
  k_weff<<<128, 256, 0, stream>>>(W_f, W_lin, W_eff);
  k_beff_pad<<<1, 128, 0, stream>>>(W_f, b_lin, b_f, pad, W_eff, b_eff, pad_vec);
  k_wsplit<<<64, 256, 0, stream>>>(W_eff, Whi, Wlo);

  // 2) xg = x @ W_g^T + b_g  (written straight into d_out's x region)
  wmma_gemm_f32<<<(TT * EE * BB) / 128, 256, 0, stream>>>(
      x, W_g, 128, b_g, out_x);

  // 3) xmean over T
  k_xmean<<<EBD / 256, 256, 0, stream>>>(out_x, xmean);

  // 4) xadd[e,b] = xmean[e,b] @ W2^T + b_eff   (W2 = W_eff[:,128:256])
  wmma_gemm_f32<<<(EE * BB) / 128, 256, 0, stream>>>(
      xmean, W_eff + 128, 256, b_eff, xadd);

  // 5) scatter table: (n0,n1) -> e  (else pad)
  k_eidx_fill<<<(NN * NN + 255) / 256, 256, 0, stream>>>(eidx);
  k_eidx_scatter<<<(EE + 255) / 256, 256, 0, stream>>>(nodes, eidx);

  // 6) feats_out = feats @ W1^T + {xadd | pad_vec}  (split-bf16 WMMA)
  wmma_gemm_bf16x3<<<FEAT_ROWS / 128, 256, 0, stream>>>(
      feats, Whi, Wlo, eidx, xadd, pad_vec, out_feats);

  // 7) x_out += gathered feats rows
  k_final<<<TEBD / 256, 256, 0, stream>>>(nodes, out_feats, out_x);
}
